// Attention_57269093925300
// MI455X (gfx1250) — compile-verified, matrix operand order checked by reading, not run on hardware
//
#include <hip/hip_runtime.h>
#include <hip/hip_bf16.h>
#include <stdint.h>

// ---------------- problem constants (from reference) ----------------
#define DIM       1024
#define HEAD_DIM  64
#define NUM_HEADS 16
#define INNER     1024        // HEAD_DIM * NUM_HEADS
#define NQKV      3072        // 3 * INNER
#define SEQ       2048
#define BATCH     4
#define ROWS      (BATCH * SEQ)   // 8192
#define ATTN_SCALE 0.125f         // 64^-0.5

// ---------------- WMMA types ----------------
typedef __attribute__((ext_vector_type(16))) __bf16   v16bf;
typedef __attribute__((ext_vector_type(8)))  float    v8f;
typedef __attribute__((ext_vector_type(4)))  uint32_t v4u;

static __device__ __forceinline__ v8f wmma_bf16(v16bf a, v16bf b, v8f c) {
    // D(f32 16x16) = A(16x32 bf16) * B(32x16 bf16) + C
    return __builtin_amdgcn_wmma_f32_16x16x32_bf16(
        /*neg_a=*/false, a, /*neg_b=*/false, b,
        /*c_mod=*/(short)0, c, /*reuse_a=*/false, /*reuse_b=*/false);
}

// round-to-nearest-even f32 -> bf16 (bit pattern as ushort)
static __device__ __forceinline__ uint16_t f2bf(float f) {
    union { float f; uint32_t u; } c; c.f = f;
    uint32_t u = c.u + 0x7FFFu + ((c.u >> 16) & 1u);
    return (uint16_t)(u >> 16);
}

union BfPack { v4u u[2]; v16bf v; };

// Load one 16x32 bf16 A/B fragment for this lane.
// Per ISA 7.12.2: lane(m=lane&15, half=lane>>4) needs K = [8h,8h+8) and
// [16+8h,16+8h+8) -> two contiguous 16-byte loads from the lane's row.
static __device__ __forceinline__ v16bf load_frag(const uint16_t* rowbase, int half) {
    BfPack p;
    p.u[0] = *(const v4u*)(rowbase + 8 * half);
    p.u[1] = *(const v4u*)(rowbase + 16 + 8 * half);
    return p.v;
}

// reductions across the 16 lanes sharing a C-layout row (xor masks < 16 stay
// inside each half of the wave32)
static __device__ __forceinline__ float red_max16(float v) {
    v = fmaxf(v, __shfl_xor(v, 1, 32));
    v = fmaxf(v, __shfl_xor(v, 2, 32));
    v = fmaxf(v, __shfl_xor(v, 4, 32));
    v = fmaxf(v, __shfl_xor(v, 8, 32));
    return v;
}
static __device__ __forceinline__ float red_sum16(float v) {
    v += __shfl_xor(v, 1, 32);
    v += __shfl_xor(v, 2, 32);
    v += __shfl_xor(v, 4, 32);
    v += __shfl_xor(v, 8, 32);
    return v;
}

// ---- CDNA5 async-to-LDS staging (ASYNCcnt path, CDNA5 ISA) --------
// Copy one 16-byte segment per thread: 256 threads cover a 64-row x 32-elem
// bf16 tile (4 KB). VDST VGPR = LDS byte offset (low 32 bits of the generic
// pointer to a __shared__ object per the aperture mapping), VADDR = 64-bit
// global address, GV mode.
static __device__ __forceinline__ void async_b_tile(const uint16_t* __restrict__ src,
                                                    uint16_t* __restrict__ dst,
                                                    int K) {
    const int tid = threadIdx.x;
    const int row = tid >> 2;        // 0..63
    const int seg = tid & 3;         // 4 x 16B per row
    const uint16_t* g = src + (size_t)row * K + seg * 8;
    uint32_t lds = (uint32_t)(uintptr_t)(dst + row * 32 + seg * 8);
    asm volatile("global_load_async_to_lds_b128 %0, %1, off"
                 :: "v"(lds), "v"(g)
                 : "memory");
}

// Shared WMMA GEMM mainloop: A rows direct from global (no cross-wave reuse),
// B tile staged through double-buffered LDS with async loads, prefetching the
// next k-chunk while WMMAs consume the current one.
static __device__ __forceinline__ void gemm_mainloop(const uint16_t* __restrict__ arow,
                                                     const uint16_t* __restrict__ bt,
                                                     uint16_t* __restrict__ shm, // [2][64][32]
                                                     v8f acc[4], int K,
                                                     int half, int lm) {
    int buf = 0;
    async_b_tile(bt, shm, K);                 // k-chunk 0 -> buf 0
    for (int k0 = 0; k0 < K; k0 += 32) {
        if (k0 + 32 < K) {
            async_b_tile(bt + (k0 + 32), shm + (buf ^ 1) * (64 * 32), K);
            asm volatile("s_wait_asynccnt 0x1" ::: "memory"); // oldest tile landed
        } else {
            asm volatile("s_wait_asynccnt 0x0" ::: "memory");
        }
        __syncthreads();                      // tile visible to all waves

        v16bf a = load_frag(arow + k0, half);
        const uint16_t* bb = shm + buf * (64 * 32);
#pragma unroll
        for (int t = 0; t < 4; ++t)
            acc[t] = wmma_bf16(a, load_frag(bb + (t * 16 + lm) * 32, half), acc[t]);

        __syncthreads();                      // reads done before buffer reuse
        buf ^= 1;
    }
}

// ---------------- conversion kernels ----------------
__global__ void cvt_bf16(const float* __restrict__ in, uint16_t* __restrict__ out, int n) {
    int i = blockIdx.x * blockDim.x + threadIdx.x;
    if (i < n) out[i] = f2bf(in[i]);
}

// in[K][N] (row-major) -> out[N][K] bf16 (GEMM B columns become contiguous rows)
__global__ void cvt_bf16_T(const float* __restrict__ in, uint16_t* __restrict__ out, int K, int N) {
    int i = blockIdx.x * blockDim.x + threadIdx.x;
    if (i < K * N) {
        int n = i / K, k = i - n * K;
        out[i] = f2bf(in[(size_t)k * N + n]);
    }
}

// ---------------- QKV GEMM: [8192x1024]bf16 @ [1024x3072] -> scatter q/k/vT ----
// block = 256 thr (8 waves); block tile = 128(M) x 64(N); wave tile = 16x64.
__global__ void qkv_gemm(const uint16_t* __restrict__ xb, const uint16_t* __restrict__ wT,
                         uint16_t* __restrict__ qb, uint16_t* __restrict__ kb,
                         uint16_t* __restrict__ vtb) {
    __shared__ __align__(16) uint16_t bshm[2][64][32];

    const int lane = threadIdx.x & 31;
    const int wave = threadIdx.x >> 5;
    const int half = lane >> 4;
    const int lm   = lane & 15;

    const int arow_idx = blockIdx.y * 128 + wave * 16 + lm;   // A-layout: m = lane&15
    const uint16_t* arow = xb + (size_t)arow_idx * DIM;
    const int colbase = blockIdx.x * 64;

    v8f acc[4] = {};
    gemm_mainloop(arow, wT + (size_t)colbase * DIM, &bshm[0][0][0], acc, DIM, half, lm);

    // epilogue: C-layout row = r + 8*half, col = t*16 + lm. Scatter into
    // q[b,h,n,d], k[b,h,n,d], vT[b,h,d,n] as bf16.
    const int rbase = blockIdx.y * 128 + wave * 16 + 8 * half;
#pragma unroll
    for (int t = 0; t < 4; ++t) {
        int c     = colbase + t * 16 + lm;
        int part  = c >> 10;           // 0=q 1=k 2=v
        int inner = c & 1023;
        int head  = inner >> 6;
        int d     = inner & 63;
#pragma unroll
        for (int r = 0; r < 8; ++r) {
            int grow = rbase + r;
            int b = grow >> 11, n = grow & 2047;
            size_t bh = (size_t)(b * NUM_HEADS + head);
            uint16_t val = f2bf(acc[t][r]);
            if (part == 0)      qb [(bh * SEQ + n) * HEAD_DIM + d] = val;
            else if (part == 1) kb [(bh * SEQ + n) * HEAD_DIM + d] = val;
            else                vtb[(bh * HEAD_DIM + d) * SEQ + n] = val;
        }
    }
}

// ---------------- flash attention: one (b,h) per blockIdx.y, 128 q-rows/block --
__global__ void attn_fwd(const uint16_t* __restrict__ qb, const uint16_t* __restrict__ kb,
                         const uint16_t* __restrict__ vtb, uint16_t* __restrict__ ctx) {
    __shared__ __align__(16) uint16_t pshm[8][16][32];   // per-wave P tile (bf16)

    const int lane = threadIdx.x & 31;
    const int wave = threadIdx.x >> 5;
    const int half = lane >> 4;
    const int lm   = lane & 15;

    const int bh = blockIdx.y;
    const uint16_t* Qh = qb  + (size_t)bh * SEQ * HEAD_DIM;
    const uint16_t* Kh = kb  + (size_t)bh * SEQ * HEAD_DIM;
    const uint16_t* Vt = vtb + (size_t)bh * HEAD_DIM * SEQ;

    const int rowbase = blockIdx.x * 128 + wave * 16;
    const uint16_t* qrow = Qh + (size_t)(rowbase + lm) * HEAD_DIM;
    const v16bf q0 = load_frag(qrow, half);        // d = 0..31
    const v16bf q1 = load_frag(qrow + 32, half);   // d = 32..63

    float mrow[8], lrow[8];
    v8f o[4] = {};
#pragma unroll
    for (int r = 0; r < 8; ++r) { mrow[r] = -1e30f; lrow[r] = 0.0f; }

    for (int j0 = 0; j0 < SEQ; j0 += 32) {
        // ---- S = Q @ K^T for 32 keys (two 16-key C tiles) ----
        const uint16_t* krow0 = Kh + (size_t)(j0 + lm) * HEAD_DIM;
        const uint16_t* krow1 = Kh + (size_t)(j0 + 16 + lm) * HEAD_DIM;
        v8f s0 = {}, s1 = {};
        s0 = wmma_bf16(q0, load_frag(krow0, half), s0);
        s0 = wmma_bf16(q1, load_frag(krow0 + 32, half), s0);
        s1 = wmma_bf16(q0, load_frag(krow1, half), s1);
        s1 = wmma_bf16(q1, load_frag(krow1 + 32, half), s1);

        // ---- online softmax (per C-layout row r + 8*half) ----
        float p0[8], p1[8], alpha[8];
#pragma unroll
        for (int r = 0; r < 8; ++r) {
            float a  = s0[r] * ATTN_SCALE;
            float bv = s1[r] * ATTN_SCALE;
            float mx = red_max16(fmaxf(a, bv));
            float mn = fmaxf(mrow[r], mx);
            alpha[r] = __expf(mrow[r] - mn);
            p0[r]    = __expf(a  - mn);
            p1[r]    = __expf(bv - mn);
            lrow[r]  = lrow[r] * alpha[r] + red_sum16(p0[r] + p1[r]);
            mrow[r]  = mn;
        }
#pragma unroll
        for (int t = 0; t < 4; ++t)
#pragma unroll
            for (int r = 0; r < 8; ++r) o[t][r] *= alpha[r];

        // ---- C-layout P (col-per-lane) -> A-layout (row-per-lane) via LDS ----
        __syncthreads();   // WAR vs previous iteration's reads
#pragma unroll
        for (int r = 0; r < 8; ++r) {
            pshm[wave][r + 8 * half][lm]      = f2bf(p0[r]);
            pshm[wave][r + 8 * half][16 + lm] = f2bf(p1[r]);
        }
        __syncthreads();   // RAW before fragment reload
        v16bf pa = load_frag(&pshm[wave][lm][0], half);

        // ---- O += P @ V ; V^T[d][n] makes B-operand rows contiguous ----
#pragma unroll
        for (int t = 0; t < 4; ++t) {
            const uint16_t* vrow = Vt + (size_t)(t * 16 + lm) * SEQ + j0;
            o[t] = wmma_bf16(pa, load_frag(vrow, half), o[t]);
        }
    }

    // ---- normalize and write context as bf16 [b, n, h*64 + d] ----
    const int head = bh & (NUM_HEADS - 1);
    const int b    = bh >> 4;
#pragma unroll
    for (int t = 0; t < 4; ++t) {
        int d = t * 16 + lm;
#pragma unroll
        for (int r = 0; r < 8; ++r) {
            int n = rowbase + r + 8 * half;
            float val = o[t][r] / lrow[r];
            ctx[((size_t)(b * SEQ + n)) * INNER + head * HEAD_DIM + d] = f2bf(val);
        }
    }
}

// ---------------- output projection: ctx[8192x1024]bf16 @ wProjT + bias -> f32 --
__global__ void proj_gemm(const uint16_t* __restrict__ ctx, const uint16_t* __restrict__ wT,
                          const float* __restrict__ bias, float* __restrict__ out) {
    __shared__ __align__(16) uint16_t bshm[2][64][32];

    const int lane = threadIdx.x & 31;
    const int wave = threadIdx.x >> 5;
    const int half = lane >> 4;
    const int lm   = lane & 15;

    const int arow_idx = blockIdx.y * 128 + wave * 16 + lm;
    const uint16_t* arow = ctx + (size_t)arow_idx * INNER;
    const int colbase = blockIdx.x * 64;

    v8f acc[4] = {};
    gemm_mainloop(arow, wT + (size_t)colbase * INNER, &bshm[0][0][0], acc, INNER, half, lm);

    const int rbase = blockIdx.y * 128 + wave * 16 + 8 * half;
#pragma unroll
    for (int t = 0; t < 4; ++t) {
        int c = colbase + t * 16 + lm;
        float bv = bias[c];
#pragma unroll
        for (int r = 0; r < 8; ++r) {
            int grow = rbase + r;
            out[(size_t)grow * DIM + c] = acc[t][r] + bv;
        }
    }
}

// ---------------- launcher ----------------
extern "C" void kernel_launch(void* const* d_in, const int* in_sizes, int n_in,
                              void* d_out, int out_size, void* d_ws, size_t ws_size,
                              hipStream_t stream) {
    const float* x      = (const float*)d_in[0];   // [4,2048,1024]
    const float* w_qkv  = (const float*)d_in[1];   // [1024,3072]
    const float* w_proj = (const float*)d_in[2];   // [1024,1024]
    const float* b_proj = (const float*)d_in[3];   // [1024]
    float* out = (float*)d_out;                    // [4,2048,1024]

    uint8_t* ws = (uint8_t*)d_ws;
    size_t off = 0;
    uint16_t* xb     = (uint16_t*)(ws + off); off += (size_t)ROWS * DIM * 2;           // 16 MB
    uint16_t* wqkvT  = (uint16_t*)(ws + off); off += (size_t)NQKV * DIM * 2;           //  6 MB
    uint16_t* wprojT = (uint16_t*)(ws + off); off += (size_t)DIM * INNER * 2;          //  2 MB
    uint16_t* qb     = (uint16_t*)(ws + off); off += (size_t)BATCH * NUM_HEADS * SEQ * HEAD_DIM * 2; // 16 MB
    uint16_t* kbuf   = (uint16_t*)(ws + off); off += (size_t)BATCH * NUM_HEADS * SEQ * HEAD_DIM * 2; // 16 MB
    uint16_t* vtb    = (uint16_t*)(ws + off); off += (size_t)BATCH * NUM_HEADS * HEAD_DIM * SEQ * 2; // 16 MB
    uint16_t* ctxb   = (uint16_t*)(ws + off); off += (size_t)ROWS * INNER * 2;         // 16 MB
    (void)ws_size; (void)in_sizes; (void)n_in; (void)out_size;

    // 1) precision conversion (+ weight transposes for contiguous B operands)
    cvt_bf16  <<<(ROWS * DIM + 255) / 256, 256, 0, stream>>>(x, xb, ROWS * DIM);
    cvt_bf16_T<<<(DIM * NQKV + 255) / 256, 256, 0, stream>>>(w_qkv, wqkvT, DIM, NQKV);
    cvt_bf16_T<<<(INNER * DIM + 255) / 256, 256, 0, stream>>>(w_proj, wprojT, INNER, DIM);

    // 2) QKV projection -> q[b,h,n,d], k[b,h,n,d], v^T[b,h,d,n]
    qkv_gemm<<<dim3(NQKV / 64, ROWS / 128), 256, 0, stream>>>(xb, wqkvT, qb, kbuf, vtb);

    // 3) flash attention per (b,h)
    attn_fwd<<<dim3(SEQ / 128, BATCH * NUM_HEADS), 256, 0, stream>>>(qb, kbuf, vtb, ctxb);

    // 4) output projection + bias (f32 out)
    proj_gemm<<<dim3(DIM / 64, ROWS / 128), 256, 0, stream>>>(ctxb, wprojT, b_proj, out);
}
